// CrossAttention_12051678233398
// MI455X (gfx1250) — matrix operand order repaired and checked by reading, not compiled, not run on hardware
//
#include <hip/hip_runtime.h>

typedef unsigned short u16;
typedef __attribute__((ext_vector_type(16))) __bf16          v16bf;
typedef __attribute__((ext_vector_type(8)))  float           v8f;
typedef __attribute__((ext_vector_type(8)))  unsigned int    v8u;
typedef __attribute__((ext_vector_type(4)))  unsigned int    v4u;
typedef __attribute__((ext_vector_type(8)))  int             v8i;
typedef __attribute__((ext_vector_type(4)))  int             v4i;

#define DEV __device__ __forceinline__

// ---- Tensor Data Mover availability (probe-verified builtins, arity differs
// ---- between ROCm 7.2 (5 args) and therock headers (6 args)) ----------------
#if defined(__has_builtin)
#  if __has_builtin(__builtin_amdgcn_tensor_load_to_lds) && \
      __has_builtin(__builtin_amdgcn_s_wait_tensorcnt)
#    define TDM_OK 1
#  endif
#endif
#ifndef TDM_OK
#  define TDM_OK 0
#endif
#if TDM_OK
#  if __has_include(<hip/amd_detail/amd_gfx1250_TDM.h>)
#    define TDM_SIX_ARGS 1
#  else
#    define TDM_SIX_ARGS 0
#  endif
#endif

// fp32 -> bf16 round-to-nearest-even
DEV u16 f2bf(float f) {
  unsigned int u = __builtin_bit_cast(unsigned int, f);
  u += 0x7fffu + ((u >> 16) & 1u);
  return (u16)(u >> 16);
}

DEV v8f wmma_bf16(v16bf a, v16bf b, v8f c) {
  // D = A(16x32 bf16) * B(32x16 bf16) + C(16x16 f32)
  return __builtin_amdgcn_wmma_f32_16x16x32_bf16(
      /*neg_a=*/false, a, /*neg_b=*/false, b,
      /*c_mod=*/(short)0, c, /*reuse_a=*/false, /*reuse_b=*/false);
}

// A-operand (16x32 bf16) gather from LDS tile T[row-major, stride ld].
// Lanes 0-15: M=lane, K in {0..7,16..23}; lanes 16-31: M=lane-16, K in {8..15,24..31}.
DEV v16bf frag_a_g(const u16* T, int ld, int row, int kofs, int lane) {
  const int m  = lane & 15;
  const int kb = (lane & 16) ? 8 : 0;
  const u16* base = T + (size_t)(row + m) * ld + kofs + kb;
  v8u u;
#pragma unroll
  for (int j = 0; j < 8; ++j) {
    const int e = 2 * j;
    const int k = e + ((e >= 8) ? 8 : 0);
    u[j] = *(const unsigned int*)(base + k);
  }
  return __builtin_bit_cast(v16bf, u);
}

// B-operand (32x16 bf16): B(k,n) = T[row+n][kofs+k], K contiguous in columns.
// Lanes 0-15: N=lane, K in {0..7,16..23}; lanes 16-31: N=lane-16, K in {8..15,24..31}.
DEV v16bf frag_b_g(const u16* T, int ld, int row, int kofs, int lane) {
  const int n  = lane & 15;
  const int kb = (lane & 16) ? 8 : 0;
  const u16* base = T + (size_t)(row + n) * ld + kofs + kb;
  v8u u;
#pragma unroll
  for (int j = 0; j < 8; ++j) u[j] = *(const unsigned int*)(base + 2 * j + ((2 * j >= 8) ? 8 : 0));
  return __builtin_bit_cast(v16bf, u);
}

#if TDM_OK
// Issue a TDM load of a 2D bf16 tile (rows x cols, row stride in elements)
// from global memory into LDS at byte offset lds_off. D# packing per ISA
// CDNA5 ISA (2D: groups 2/3 zero).
DEV void tdm_load_2d_bf16(unsigned lds_off, const u16* gsrc,
                          int rows, int cols, int stride_elems) {
  const unsigned long long ga = (unsigned long long)gsrc;
  v4u g0;
  g0[0] = 1u;                                            // count=1, user mode
  g0[1] = lds_off;                                       // lds_addr (bytes)
  g0[2] = (unsigned)(ga & 0xffffffffull);                // global_addr[31:0]
  g0[3] = (unsigned)((ga >> 32) & 0x01ffffffull)         // global_addr[56:32]
          | 0x80000000u;                                 // type=2 ("image")
  v8i g1;
  g1[0] = (int)(1u << 16);                               // data_size=1 (2 bytes)
  g1[1] = (int)(((unsigned)cols & 0xffffu) << 16);       // tensor_dim0[15:0]
  g1[2] = (int)((((unsigned)cols >> 16) & 0xffffu) |
                (((unsigned)rows & 0xffffu) << 16));     // td0[31:16] | td1[15:0]
  g1[3] = (int)((((unsigned)rows >> 16) & 0xffffu) |
                (((unsigned)cols & 0xffffu) << 16));     // td1[31:16] | tile_dim0
  g1[4] = (int)((unsigned)rows & 0xffffu);               // tile_dim1 (tile_dim2=0)
  g1[5] = (int)stride_elems;                             // tensor_dim0_stride lo32
  g1[6] = 0;                                             // stride0 hi | stride1 lo
  g1[7] = 0;                                             // stride1 hi
  const v4i z4 = (v4i){0, 0, 0, 0};
#if TDM_SIX_ARGS
  const v8i z8 = (v8i){0, 0, 0, 0, 0, 0, 0, 0};
  __builtin_amdgcn_tensor_load_to_lds(g0, g1, z4, z4, z8, 0);
#else
  __builtin_amdgcn_tensor_load_to_lds(g0, g1, z4, z4, 0);
#endif
}
#endif

DEV void wait_tdm() {
#if TDM_OK
  __builtin_amdgcn_s_wait_tensorcnt(0);
#endif
}

// Vectorized fp32 -> bf16 convert: float4 in, packed 4 x bf16 out.
__global__ void cvt4_f32_bf16_kernel(const float* __restrict__ in,
                                     u16* __restrict__ out, int n4) {
  int i = blockIdx.x * blockDim.x + threadIdx.x;
  if (i < n4) {
    float4 v = ((const float4*)in)[i];
    uint2 o;
    o.x = (unsigned)f2bf(v.x) | ((unsigned)f2bf(v.y) << 16);
    o.y = (unsigned)f2bf(v.z) | ((unsigned)f2bf(v.w) << 16);
    ((uint2*)out)[i] = o;
  }
}

// C[M,N] = A[M,K] * W[N,K]^T, 128x128 block tile, 4 waves (2x2), each wave a
// 64x64 accumulator (4x4 WMMA tiles -> 32 LDS-FLOP/byte). K-tiles (128x32)
// double-buffered in LDS; both operands staged by the Tensor Data Mover when
// available so the k+1 transfer overlaps the 16 WMMAs of step k.
template <bool OUT_F32>
__global__ void __launch_bounds__(128) gemm_wmma_kernel(
    const u16* __restrict__ A, const u16* __restrict__ W,
    const float* __restrict__ bias, void* __restrict__ Outp,
    int M, int N, int K) {
  __shared__ __align__(16) u16 As[2][128][32];
  __shared__ __align__(16) u16 Ws[2][128][32];

  const int tid  = threadIdx.x;
  const int lane = tid & 31;
  const int w    = tid >> 5;
  const int wm   = w >> 1;
  const int wn   = w & 1;
  const int m0   = blockIdx.x * 128;
  const int n0   = blockIdx.y * 128;
  const int nk   = K >> 5;

  v8f acc[4][4];
#pragma unroll
  for (int i = 0; i < 4; ++i)
#pragma unroll
    for (int j = 0; j < 4; ++j) acc[i][j] = (v8f){0.f,0.f,0.f,0.f,0.f,0.f,0.f,0.f};

  auto fill = [&](int buf, int kt) {
    const int k0 = kt << 5;
#if TDM_OK
    if (w == 0)
      tdm_load_2d_bf16((unsigned)(unsigned long long)&As[buf][0][0],
                       A + (size_t)m0 * K + k0, 128, 32, K);
    if (w == 1)
      tdm_load_2d_bf16((unsigned)(unsigned long long)&Ws[buf][0][0],
                       W + (size_t)n0 * K + k0, 128, 32, K);
#else
#pragma unroll
    for (int i = 0; i < 4; ++i) {            // 512 slots: 128 rows x 4 uint4
      int s = tid + i * 128;
      int row = s >> 2, q = s & 3;
      *(uint4*)&As[buf][row][q * 8] =
          *(const uint4*)(A + (size_t)(m0 + row) * K + k0 + q * 8);
      *(uint4*)&Ws[buf][row][q * 8] =
          *(const uint4*)(W + (size_t)(n0 + row) * K + k0 + q * 8);
    }
    if (kt + 1 < nk) {
      __builtin_prefetch(A + (size_t)(m0 + (tid >> 2)) * K + k0 + 32, 0, 1);
      __builtin_prefetch(W + (size_t)(n0 + (tid >> 2)) * K + k0 + 32, 0, 1);
    }
#endif
  };

  fill(0, 0);
  wait_tdm();
  __syncthreads();

  for (int kt = 0; kt < nk; ++kt) {
    const int cur = kt & 1;
    if (kt + 1 < nk) fill(cur ^ 1, kt + 1);  // overlap transfer with WMMA

    v16bf a[4], b[4];
#pragma unroll
    for (int t = 0; t < 4; ++t) {
      a[t] = frag_a_g(&As[cur][0][0], 32, wm * 64 + t * 16, 0, lane);
      b[t] = frag_b_g(&Ws[cur][0][0], 32, wn * 64 + t * 16, 0, lane);
    }
#pragma unroll
    for (int mt = 0; mt < 4; ++mt)
#pragma unroll
      for (int nt = 0; nt < 4; ++nt)
        acc[mt][nt] = wmma_bf16(a[mt], b[nt], acc[mt][nt]);

    wait_tdm();        // issuer waits for next buffer; no-op for other waves
    __syncthreads();
  }

  // C/D layout: element (r, lane) -> row = r + 8*(lane>=16), col = lane&15
  const int mhi = (lane & 16) >> 1;
  const int nn  = lane & 15;
#pragma unroll
  for (int mt = 0; mt < 4; ++mt)
#pragma unroll
    for (int nt = 0; nt < 4; ++nt)
#pragma unroll
      for (int r = 0; r < 8; ++r) {
        int row = m0 + wm * 64 + mt * 16 + r + mhi;
        int col = n0 + wn * 64 + nt * 16 + nn;
        float v = acc[mt][nt][r];
        if (OUT_F32)
          ((float*)Outp)[(size_t)row * N + col] = v + bias[col];
        else
          ((u16*)Outp)[(size_t)row * N + col] = f2bf(v);
      }
}

// Flash attention over one (b, h, 64-query tile). 4 waves; wave w owns 16
// query rows. Q/K streamed via TDM; V staged transposed (Vt[d][key]) so the
// P*V B-operand is a contiguous-K gather (ds_load_b128, no scalar u16s).
__global__ void __launch_bounds__(128) attn_fwd_kernel(
    const u16* __restrict__ Qp, const u16* __restrict__ Kp,
    const u16* __restrict__ Vp, u16* __restrict__ Xb) {
  constexpr int NQc = 1024, NKc = 1024, Cc = 768, Hc = 12, Dc = 64;
  __shared__ __align__(16) u16 Qs[64][64];
  __shared__ __align__(16) u16 Ks[64][64];
  __shared__ __align__(16) u16 Vt[64][72];   // [d][key], padded stride
  __shared__ __align__(16) u16 Ps[4][16][32];

  const int tid  = threadIdx.x;
  const int lane = tid & 31;
  const int wq   = tid >> 5;

  const int nqt = NQc / 64;                  // 16
  const int qt  = blockIdx.x % nqt;
  const int h   = (blockIdx.x / nqt) % Hc;
  const int b   = blockIdx.x / (nqt * Hc);
  const int q0  = qt * 64;

  const size_t qbase = (size_t)b * NQc * Cc + (size_t)h * Dc;
  const size_t kbase = (size_t)b * NKc * Cc + (size_t)h * Dc;

#if TDM_OK
  if (wq == 0)
    tdm_load_2d_bf16((unsigned)(unsigned long long)&Qs[0][0],
                     Qp + qbase + (size_t)q0 * Cc, 64, 64, Cc);
  wait_tdm();
#else
#pragma unroll
  for (int i = 0; i < 4; ++i) {              // 512 slots: 64 rows x 8 uint4
    int s = tid + i * 128;
    int row = s >> 3, q = s & 7;
    *(uint4*)&Qs[row][q * 8] =
        *(const uint4*)(Qp + qbase + (size_t)(q0 + row) * Cc + q * 8);
  }
#endif
  __syncthreads();

  v16bf aq[2];
  aq[0] = frag_a_g(&Qs[0][0], 64, wq * 16, 0,  lane);
  aq[1] = frag_a_g(&Qs[0][0], 64, wq * 16, 32, lane);

  v8f O[4];
#pragma unroll
  for (int dt = 0; dt < 4; ++dt) O[dt] = (v8f){0.f,0.f,0.f,0.f,0.f,0.f,0.f,0.f};
  float mrow[8], lrow[8];
#pragma unroll
  for (int r = 0; r < 8; ++r) { mrow[r] = -1.0e30f; lrow[r] = 0.f; }

  const float scale = 0.125f;                // 1/sqrt(64)
  const int mhi = (lane & 16) >> 1;          // 0 or 8
  const int nn  = lane & 15;

  for (int kc = 0; kc < NKc / 64; ++kc) {
    __syncthreads();                         // previous chunk fully consumed
    const int kr0 = kc * 64;
#if TDM_OK
    if (wq == 0)
      tdm_load_2d_bf16((unsigned)(unsigned long long)&Ks[0][0],
                       Kp + kbase + (size_t)kr0 * Cc, 64, 64, Cc);
#else
#pragma unroll
    for (int i = 0; i < 4; ++i) {
      int s = tid + i * 128;
      int row = s >> 3, q = s & 7;
      *(uint4*)&Ks[row][q * 8] =
          *(const uint4*)(Kp + kbase + (size_t)(kr0 + row) * Cc + q * 8);
    }
#endif
    // V: load 16B rows, scatter transposed into Vt[d][key]
#pragma unroll
    for (int i = 0; i < 4; ++i) {
      int s = tid + i * 128;
      int row = s >> 3, q = s & 7;
      uint4 vv = *(const uint4*)(Vp + kbase + (size_t)(kr0 + row) * Cc + q * 8);
      const u16* e = (const u16*)&vv;
#pragma unroll
      for (int c = 0; c < 8; ++c) Vt[q * 8 + c][row] = e[c];
    }
    wait_tdm();
    __syncthreads();

#pragma unroll
    for (int nc = 0; nc < 2; ++nc) {         // two 32-key sub-chunks
      // hoist all K fragments so LDS latency overlaps the WMMA group
      v16bf bk[2][2];
#pragma unroll
      for (int kd = 0; kd < 2; ++kd) {
        bk[kd][0] = frag_b_g(&Ks[0][0], 64, nc * 32,      kd * 32, lane);
        bk[kd][1] = frag_b_g(&Ks[0][0], 64, nc * 32 + 16, kd * 32, lane);
      }
      v8f s0 = (v8f){0.f,0.f,0.f,0.f,0.f,0.f,0.f,0.f};
      v8f s1 = s0;
#pragma unroll
      for (int kd = 0; kd < 2; ++kd) {       // D=64 -> 2 WMMA K-steps
        s0 = wmma_bf16(aq[kd], bk[kd][0], s0);
        s1 = wmma_bf16(aq[kd], bk[kd][1], s1);
      }
      // V fragments: independent of softmax, issue early
      v16bf bv[4];
#pragma unroll
      for (int dt = 0; dt < 4; ++dt)
        bv[dt] = frag_b_g(&Vt[0][0], 72, dt * 16, nc * 32, lane);

#pragma unroll
      for (int r = 0; r < 8; ++r) {
        float x0 = s0[r] * scale;
        float x1 = s1[r] * scale;
        float rm = fmaxf(x0, x1);
        rm = fmaxf(rm, __shfl_xor(rm, 1, 32));
        rm = fmaxf(rm, __shfl_xor(rm, 2, 32));
        rm = fmaxf(rm, __shfl_xor(rm, 4, 32));
        rm = fmaxf(rm, __shfl_xor(rm, 8, 32));
        float mn = fmaxf(mrow[r], rm);
        float al = __expf(mrow[r] - mn);
        mrow[r] = mn;
        x0 = __expf(x0 - mn);
        x1 = __expf(x1 - mn);
        float rs = x0 + x1;
        rs += __shfl_xor(rs, 1, 32);
        rs += __shfl_xor(rs, 2, 32);
        rs += __shfl_xor(rs, 4, 32);
        rs += __shfl_xor(rs, 8, 32);
        lrow[r] = lrow[r] * al + rs;
#pragma unroll
        for (int dt = 0; dt < 4; ++dt) O[dt][r] *= al;
        Ps[wq][r + mhi][nn]      = f2bf(x0);   // restage P in A-operand shape
        Ps[wq][r + mhi][16 + nn] = f2bf(x1);
      }
      // DS ops from the same wave complete in order: read Ps back directly.
      v16bf ap = frag_a_g(&Ps[wq][0][0], 32, 0, 0, lane);
#pragma unroll
      for (int dt = 0; dt < 4; ++dt)
        O[dt] = wmma_bf16(ap, bv[dt], O[dt]);
    }
  }

#pragma unroll
  for (int dt = 0; dt < 4; ++dt)
#pragma unroll
    for (int r = 0; r < 8; ++r) {
      float o = O[dt][r] / lrow[r];
      size_t idx = qbase + (size_t)(q0 + wq * 16 + r + mhi) * Cc + dt * 16 + nn;
      Xb[idx] = f2bf(o);
    }
}

extern "C" void kernel_launch(void* const* d_in, const int* in_sizes, int n_in,
                              void* d_out, int out_size, void* d_ws, size_t ws_size,
                              hipStream_t stream) {
  (void)in_sizes; (void)n_in; (void)out_size; (void)ws_size;
  constexpr int B = 16, NQ = 1024, C = 768, H = 12;
  constexpr int M = B * NQ;  // 16384

  const float* query = (const float*)d_in[0];
  const float* key   = (const float*)d_in[1];
  const float* value = (const float*)d_in[2];
  const float* Wq    = (const float*)d_in[3];
  const float* Wk    = (const float*)d_in[4];
  const float* Wv    = (const float*)d_in[5];
  const float* Wo    = (const float*)d_in[6];
  const float* bo    = (const float*)d_in[7];

  const size_t wElems = (size_t)C * C;        // 589824
  const size_t pElems = (size_t)M * C;        // 12582912
  u16* Wqb = (u16*)d_ws;
  u16* Wkb = Wqb + wElems;
  u16* Wvb = Wkb + wElems;
  u16* Wob = Wvb + wElems;
  u16* qb  = Wob + wElems;
  u16* kb  = qb + pElems;
  u16* vb  = kb + pElems;
  u16* Qp  = vb + pElems;
  u16* Kp  = Qp + pElems;
  u16* Vp  = Kp + pElems;
  u16* Xb  = Vp + pElems;                     // total ~181 MB of d_ws

  const int w4 = (int)(wElems / 4);
  const int p4 = (int)(pElems / 4);
  cvt4_f32_bf16_kernel<<<(w4 + 255) / 256, 256, 0, stream>>>(Wq, Wqb, w4);
  cvt4_f32_bf16_kernel<<<(w4 + 255) / 256, 256, 0, stream>>>(Wk, Wkb, w4);
  cvt4_f32_bf16_kernel<<<(w4 + 255) / 256, 256, 0, stream>>>(Wv, Wvb, w4);
  cvt4_f32_bf16_kernel<<<(w4 + 255) / 256, 256, 0, stream>>>(Wo, Wob, w4);
  cvt4_f32_bf16_kernel<<<(p4 + 255) / 256, 256, 0, stream>>>(query, qb, p4);
  cvt4_f32_bf16_kernel<<<(p4 + 255) / 256, 256, 0, stream>>>(key,   kb, p4);
  cvt4_f32_bf16_kernel<<<(p4 + 255) / 256, 256, 0, stream>>>(value, vb, p4);

  dim3 gg(M / 128, C / 128, 1);               // (128, 6)
  gemm_wmma_kernel<false><<<gg, 128, 0, stream>>>(qb, Wqb, nullptr, Qp, M, C, C);
  gemm_wmma_kernel<false><<<gg, 128, 0, stream>>>(kb, Wkb, nullptr, Kp, M, C, C);
  gemm_wmma_kernel<false><<<gg, 128, 0, stream>>>(vb, Wvb, nullptr, Vp, M, C, C);

  attn_fwd_kernel<<<B * H * (NQ / 64), 128, 0, stream>>>(Qp, Kp, Vp, Xb);

  gemm_wmma_kernel<true><<<gg, 128, 0, stream>>>(Xb, Wob, bo, d_out, M, C, C);
}
